// InteractionDecoder_38139309589084
// MI455X (gfx1250) — matrix operand order repaired and checked by reading, not compiled, not run on hardware
//
#include <hip/hip_runtime.h>
#include <math.h>

#define TT   1024
#define SS   1024
#define BB   4
#define DD   512
#define HH   8
#define HD   64
#define DFFN 2048
#define MROWS (TT*BB)
#define LN_EPS 1e-5f
#define QSCALE 0.125f   /* 1/sqrt(64) */

typedef _Float16 v8h  __attribute__((ext_vector_type(8)));
typedef _Float16 v16h __attribute__((ext_vector_type(16)));
typedef float    v8f  __attribute__((ext_vector_type(8)));

// ---------------------------------------------------------------------------
// WMMA helpers (CDNA5 wave32, V_WMMA_F32_16X16X32_F16)
// ---------------------------------------------------------------------------
__device__ inline v8f wmma16(v16h a, v16h b, v8f c) {
  return __builtin_amdgcn_wmma_f32_16x16x32_f16(false, a, false, b, (short)0, c,
                                                false, false);
}

// A-matrix 16x32 f16 fragment: per lane, 8 f16 at K = (lane>>4)*8 and 8 more
// at K = 16 + (lane>>4)*8 (ISA 7.12.2 layout).  Caller passes pointer to the
// first chunk; second chunk is +16 elements.
__device__ inline v16h lda_frag(const _Float16* p) {
  v8h lo = *(const v8h*)(p);
  v8h hi = *(const v8h*)(p + 16);
  v16h r;
#pragma unroll
  for (int i = 0; i < 8; ++i) { r[i] = lo[i]; r[i + 8] = hi[i]; }
  return r;
}

// ---------------------------------------------------------------------------
// fp32 -> f16 elementwise convert
// ---------------------------------------------------------------------------
__global__ void cvt_f16(const float* __restrict__ in, _Float16* __restrict__ out,
                        int n) {
  int i = blockIdx.x * blockDim.x + threadIdx.x;
  if (i < n) out[i] = (_Float16)in[i];
}

// ---------------------------------------------------------------------------
// C[M,N] = A[M,K] (f16, row-major) x W[N,K]^T (f16, row-major) + bias, opt ReLU
// Block: 256 thr = 8 waves; block tile 128x128; wave tile 32x64 (2x4 frags).
// ---------------------------------------------------------------------------
__global__ __launch_bounds__(256) void gemm_nt(
    const _Float16* __restrict__ A, const _Float16* __restrict__ Wt,
    const float* __restrict__ bias, float* __restrict__ Cf,
    _Float16* __restrict__ Ch, int M, int N, int K, int relu) {
  const int lane   = threadIdx.x & 31;
  const int wave   = threadIdx.x >> 5;
  const int wm     = wave & 3, wn = wave >> 2;
  const int rowBase = blockIdx.y * 128 + wm * 32;
  const int colBase = blockIdx.x * 128 + wn * 64;
  const int l15    = lane & 15;
  const int half8  = (lane >> 4) * 8;
  const int half16 = (lane >> 4) * 16;

  v8f acc[2][4];
#pragma unroll
  for (int i = 0; i < 2; ++i)
#pragma unroll
    for (int j = 0; j < 4; ++j)
#pragma unroll
      for (int r = 0; r < 8; ++r) acc[i][j][r] = 0.f;

  const _Float16* A0 = A + (size_t)(rowBase + l15) * K;
  const _Float16* A1 = A0 + (size_t)16 * K;
  const _Float16* W0 = Wt + (size_t)(colBase + l15) * K;

  for (int k0 = 0; k0 < K; k0 += 32) {
    v16h a0 = lda_frag(A0 + k0 + half8);
    v16h a1 = lda_frag(A1 + k0 + half8);
#pragma unroll
    for (int j = 0; j < 4; ++j) {
      v16h b = lda_frag(W0 + (size_t)(16 * j) * K + k0 + half8);
      acc[0][j] = wmma16(a0, b, acc[0][j]);
      acc[1][j] = wmma16(a1, b, acc[1][j]);
    }
  }

#pragma unroll
  for (int i = 0; i < 2; ++i) {
#pragma unroll
    for (int j = 0; j < 4; ++j) {
      int n = colBase + 16 * j + l15;
      float bv = bias ? bias[n] : 0.f;
#pragma unroll
      for (int r = 0; r < 8; ++r) {
        int m = rowBase + 16 * i + half8 + r;
        float v = acc[i][j][r] + bv;
        if (relu) v = fmaxf(v, 0.f);
        size_t off = (size_t)m * N + n;
        if (Cf) Cf[off] = v; else Ch[off] = (_Float16)v;
      }
    }
  }
}

// ---------------------------------------------------------------------------
// Split qkv (f32 [T*B, 3D]) into Qh[B,H,T,HD]*scale, Kh[B,H,T,HD], Vt[B,H,HD,T]
// ---------------------------------------------------------------------------
__global__ void reorder_qkv(const float* __restrict__ qkv,
                            _Float16* __restrict__ Qh, _Float16* __restrict__ Kh,
                            _Float16* __restrict__ Vt) {
  int idx = blockIdx.x * blockDim.x + threadIdx.x;
  if (idx >= TT * BB * DD) return;
  int t = idx / (BB * DD);
  int rem = idx % (BB * DD);
  int b = rem / DD, e = rem % DD;
  int h = e / HD, d = e % HD;
  size_t row = (size_t)(t * BB + b) * (3 * DD);
  float q = qkv[row + e] * QSCALE;
  float k = qkv[row + DD + e];
  float v = qkv[row + 2 * DD + e];
  size_t bh = (size_t)(b * HH + h);
  Qh[(bh * TT + t) * HD + d] = (_Float16)q;
  Kh[(bh * TT + t) * HD + d] = (_Float16)k;
  Vt[(bh * HD + d) * TT + t] = (_Float16)v;
}

// ---------------------------------------------------------------------------
// Fused flash self-attention.  Grid (T/64, B*H), block 128 (4 waves).
// Each wave: 16 queries, full HD=64, online softmax over 1024 keys in 32-key
// steps.  P is staged via a 1KB/wave LDS tile to convert C-layout -> A-layout.
// Q is pre-scaled by 1/sqrt(HD).  Output Oh f16 [T*B, D] (rows t*B+b).
// ---------------------------------------------------------------------------
__global__ __launch_bounds__(128) void attn_fused(
    const _Float16* __restrict__ Qh, const _Float16* __restrict__ Kh,
    const _Float16* __restrict__ Vt, _Float16* __restrict__ Oh) {
  __shared__ __align__(16) _Float16 Pt[4][16 * 32];
  const int lane = threadIdx.x & 31;
  const int wave = threadIdx.x >> 5;
  const int bh = blockIdx.y;
  const int b = bh / HH, h = bh % HH;
  const int qt = blockIdx.x * 64 + wave * 16;
  const int l15 = lane & 15;
  const int half8 = (lane >> 4) * 8, half16 = (lane >> 4) * 16;

  const _Float16* Qb = Qh + ((size_t)bh * TT + qt) * HD;
  const _Float16* Kb = Kh + (size_t)bh * TT * HD;
  const _Float16* Vb = Vt + (size_t)bh * HD * TT;

  v16h qa0 = lda_frag(Qb + (size_t)l15 * HD + half8);
  v16h qa1 = lda_frag(Qb + (size_t)l15 * HD + 32 + half8);

  float mrun[8], lrun[8], al[8];
#pragma unroll
  for (int r = 0; r < 8; ++r) { mrun[r] = -3.0e38f; lrun[r] = 0.f; }
  v8f o[4];
#pragma unroll
  for (int j = 0; j < 4; ++j)
#pragma unroll
    for (int r = 0; r < 8; ++r) o[j][r] = 0.f;

  _Float16* P = Pt[wave];

  for (int kb = 0; kb < SS; kb += 32) {
    v8f s0, s1;
#pragma unroll
    for (int r = 0; r < 8; ++r) { s0[r] = 0.f; s1[r] = 0.f; }
    const _Float16* K0 = Kb + (size_t)(kb + l15) * HD;
    const _Float16* K1 = Kb + (size_t)(kb + 16 + l15) * HD;
    s0 = wmma16(qa0, lda_frag(K0 + half8), s0);
    s0 = wmma16(qa1, lda_frag(K0 + 32 + half8), s0);
    s1 = wmma16(qa0, lda_frag(K1 + half8), s1);
    s1 = wmma16(qa1, lda_frag(K1 + 32 + half8), s1);

    // online softmax update; rows of a C-fragment live in 16-lane halves
#pragma unroll
    for (int r = 0; r < 8; ++r) {
      float m0 = s0[r], m1 = s1[r];
#pragma unroll
      for (int msk = 1; msk < 16; msk <<= 1) {
        m0 = fmaxf(m0, __shfl_xor(m0, msk, 32));
        m1 = fmaxf(m1, __shfl_xor(m1, msk, 32));
      }
      float mn = fmaxf(mrun[r], fmaxf(m0, m1));
      float p0 = __expf(s0[r] - mn);
      float p1 = __expf(s1[r] - mn);
      float bs = p0 + p1;
#pragma unroll
      for (int msk = 1; msk < 16; msk <<= 1) bs += __shfl_xor(bs, msk, 32);
      float a = __expf(mrun[r] - mn);
      lrun[r] = lrun[r] * a + bs;
      mrun[r] = mn;
      al[r] = a;
      int m = half8 + r;
      P[m * 32 + l15] = (_Float16)p0;
      P[m * 32 + 16 + l15] = (_Float16)p1;
    }
#pragma unroll
    for (int j = 0; j < 4; ++j)
#pragma unroll
      for (int r = 0; r < 8; ++r) o[j][r] *= al[r];

    __syncthreads();
    v16h pa = lda_frag(P + l15 * 32 + half8);
#pragma unroll
    for (int j = 0; j < 4; ++j) {
      v16h vb = lda_frag(Vb + (size_t)(16 * j + l15) * TT + kb + half8);
      o[j] = wmma16(pa, vb, o[j]);
    }
    __syncthreads();
  }

#pragma unroll
  for (int j = 0; j < 4; ++j)
#pragma unroll
    for (int r = 0; r < 8; ++r) {
      int m = half8 + r;
      int t = qt + m;
      float v = o[j][r] / lrun[r];
      Oh[((size_t)(t * BB + b)) * DD + h * HD + 16 * j + l15] = (_Float16)v;
    }
}

// ---------------------------------------------------------------------------
// Sliding-window (banded, 33 keys) cross-attention; tiny FLOPs -> VALU fp32.
// q rows t*B+b; k,v rows s*B+b (all f32 [rows, D]).  Grid (T, B*H), block 64.
// ---------------------------------------------------------------------------
__global__ __launch_bounds__(64) void banded_attn(
    const float* __restrict__ q, const float* __restrict__ k,
    const float* __restrict__ v, float* __restrict__ out) {
  const int t = blockIdx.x;
  const int bh = blockIdx.y;
  const int b = bh / HH, h = bh % HH;
  const int tid = threadIdx.x;
  __shared__ float sc[33];
  __shared__ float mred, sred;

  if (tid < 33) {
    int s = t - 16 + tid;
    float acc = -1.0e30f;
    if (s >= 0 && s < SS) {
      const float* qp = q + ((size_t)(t * BB + b)) * DD + h * HD;
      const float* kp = k + ((size_t)(s * BB + b)) * DD + h * HD;
      float a = 0.f;
      for (int d = 0; d < HD; ++d) a += qp[d] * kp[d];
      acc = a * QSCALE;
    }
    sc[tid] = acc;
  }
  __syncthreads();
  if (tid == 0) {
    float m = -1.0e30f;
    for (int j = 0; j < 33; ++j) m = fmaxf(m, sc[j]);
    mred = m;
  }
  __syncthreads();
  if (tid < 33) sc[tid] = __expf(sc[tid] - mred);
  __syncthreads();
  if (tid == 0) {
    float su = 0.f;
    for (int j = 0; j < 33; ++j) su += sc[j];
    sred = 1.0f / su;
  }
  __syncthreads();
  if (tid < HD) {
    float o = 0.f;
    for (int j = 0; j < 33; ++j) {
      int s = t - 16 + j;
      if (s >= 0 && s < SS)
        o += sc[j] * v[((size_t)(s * BB + b)) * DD + h * HD + tid];
    }
    out[((size_t)(t * BB + b)) * DD + h * HD + tid] = o * sred;
  }
}

// ---------------------------------------------------------------------------
// y = LayerNorm(a + br) * g + be ; writes f32 (and optional f16 copy).
// Grid MROWS blocks, 256 threads, 2 elems/thread (D=512).
// ---------------------------------------------------------------------------
__global__ __launch_bounds__(256) void residual_ln(
    const float* __restrict__ a, const float* __restrict__ br,
    const float* __restrict__ g, const float* __restrict__ be,
    float* __restrict__ of, _Float16* __restrict__ oh) {
  __shared__ float red[256];
  const int row = blockIdx.x, tid = threadIdx.x;
  const float* pa = a + (size_t)row * DD;
  const float* pb = br + (size_t)row * DD;
  float x0 = pa[tid] + pb[tid];
  float x1 = pa[tid + 256] + pb[tid + 256];
  red[tid] = x0 + x1;
  __syncthreads();
  for (int s = 128; s > 0; s >>= 1) {
    if (tid < s) red[tid] += red[tid + s];
    __syncthreads();
  }
  float mean = red[0] * (1.0f / DD);
  __syncthreads();
  float d0 = x0 - mean, d1 = x1 - mean;
  red[tid] = d0 * d0 + d1 * d1;
  __syncthreads();
  for (int s = 128; s > 0; s >>= 1) {
    if (tid < s) red[tid] += red[tid + s];
    __syncthreads();
  }
  float inv = rsqrtf(red[0] * (1.0f / DD) + LN_EPS);
  float y0 = d0 * inv * g[tid] + be[tid];
  float y1 = d1 * inv * g[tid + 256] + be[tid + 256];
  size_t o = (size_t)row * DD + tid;
  of[o] = y0;
  of[o + 256] = y1;
  if (oh) { oh[o] = (_Float16)y0; oh[o + 256] = (_Float16)y1; }
}

// ---------------------------------------------------------------------------
extern "C" void kernel_launch(void* const* d_in, const int* in_sizes, int n_in,
                              void* d_out, int out_size, void* d_ws,
                              size_t ws_size, hipStream_t stream) {
  const float* tgt   = (const float*)d_in[0];
  const float* mem   = (const float*)d_in[1];
  const float* w_in  = (const float*)d_in[2];
  const float* b_in  = (const float*)d_in[3];
  const float* w_out = (const float*)d_in[4];
  const float* b_out = (const float*)d_in[5];
  const float* w_swq = (const float*)d_in[6];
  const float* b_swq = (const float*)d_in[7];
  const float* w_swk = (const float*)d_in[8];
  const float* b_swk = (const float*)d_in[9];
  const float* w_swv = (const float*)d_in[10];
  const float* b_swv = (const float*)d_in[11];
  const float* w_swo = (const float*)d_in[12];
  const float* b_swo = (const float*)d_in[13];
  const float* w_l1  = (const float*)d_in[14];
  const float* b_l1  = (const float*)d_in[15];
  const float* w_l2  = (const float*)d_in[16];
  const float* b_l2  = (const float*)d_in[17];
  const float* n1g = (const float*)d_in[18];
  const float* n1b = (const float*)d_in[19];
  const float* n2g = (const float*)d_in[20];
  const float* n2b = (const float*)d_in[21];
  const float* n3g = (const float*)d_in[22];
  const float* n3b = (const float*)d_in[23];

  char* p = (char*)d_ws;
  auto alloc = [&](size_t bytes) -> char* {
    char* r = p;
    p += (bytes + 255) & ~(size_t)255;
    return r;
  };

  // f16 weights
  _Float16* winH  = (_Float16*)alloc((size_t)3 * DD * DD * 2);
  _Float16* woutH = (_Float16*)alloc((size_t)DD * DD * 2);
  _Float16* wqH   = (_Float16*)alloc((size_t)DD * DD * 2);
  _Float16* wkH   = (_Float16*)alloc((size_t)DD * DD * 2);
  _Float16* wvH   = (_Float16*)alloc((size_t)DD * DD * 2);
  _Float16* woH   = (_Float16*)alloc((size_t)DD * DD * 2);
  _Float16* wl1H  = (_Float16*)alloc((size_t)DFFN * DD * 2);
  _Float16* wl2H  = (_Float16*)alloc((size_t)DD * DFFN * 2);
  // f16 activations
  _Float16* xH   = (_Float16*)alloc((size_t)MROWS * DD * 2);
  _Float16* memH = (_Float16*)alloc((size_t)MROWS * DD * 2);
  _Float16* oH   = (_Float16*)alloc((size_t)MROWS * DD * 2);
  _Float16* hH   = (_Float16*)alloc((size_t)MROWS * DFFN * 2);
  _Float16* qhH  = (_Float16*)alloc((size_t)BB * HH * TT * HD * 2);
  _Float16* khH  = (_Float16*)alloc((size_t)BB * HH * TT * HD * 2);
  _Float16* vtH  = (_Float16*)alloc((size_t)BB * HH * HD * TT * 2);
  // f32 scratch
  float* qkvF  = (float*)alloc((size_t)MROWS * 3 * DD * 4);
  float* projF = (float*)alloc((size_t)MROWS * DD * 4);
  float* x2F   = (float*)alloc((size_t)MROWS * DD * 4);
  // reuse (all strictly sequential):
  float* qswF  = qkvF;                              // after reorder, qkv dead
  float* kswF  = qkvF + (size_t)MROWS * DD;
  float* vswF  = qkvF + (size_t)2 * MROWS * DD;
  float* swoF  = (float*)qhH;                       // qh+kh contiguous 8MB
  float* x3F   = qkvF;                              // q/k/v_sw dead after band
  float* ffF   = projF;                             // swatt dead after cvt

  auto cvt = [&](const float* s, _Float16* d, int n) {
    cvt_f16<<<(n + 255) / 256, 256, 0, stream>>>(s, d, n);
  };
  auto gemm = [&](const _Float16* A, const _Float16* Wt, const float* bias,
                  float* Cf, _Float16* Ch, int M, int N, int K, int relu) {
    dim3 g(N / 128, M / 128);
    gemm_nt<<<g, 256, 0, stream>>>(A, Wt, bias, Cf, Ch, M, N, K, relu);
  };

  // 1) weight + activation conversion
  cvt(w_in, winH, 3 * DD * DD);
  cvt(w_out, woutH, DD * DD);
  cvt(w_swq, wqH, DD * DD);
  cvt(w_swk, wkH, DD * DD);
  cvt(w_swv, wvH, DD * DD);
  cvt(w_swo, woH, DD * DD);
  cvt(w_l1, wl1H, DFFN * DD);
  cvt(w_l2, wl2H, DD * DFFN);
  cvt(tgt, xH, MROWS * DD);
  cvt(mem, memH, MROWS * DD);

  // 2) self-attention
  gemm(xH, winH, b_in, qkvF, nullptr, MROWS, 3 * DD, DD, 0);
  reorder_qkv<<<(TT * BB * DD + 255) / 256, 256, 0, stream>>>(qkvF, qhH, khH,
                                                              vtH);
  attn_fused<<<dim3(TT / 64, BB * HH), 128, 0, stream>>>(qhH, khH, vtH, oH);
  gemm(oH, woutH, b_out, projF, nullptr, MROWS, DD, DD, 0);
  residual_ln<<<MROWS, 256, 0, stream>>>(tgt, projF, n1g, n1b, x2F, xH);

  // 3) sliding-window cross-attention
  gemm(xH, wqH, b_swq, qswF, nullptr, MROWS, DD, DD, 0);
  gemm(memH, wkH, b_swk, kswF, nullptr, MROWS, DD, DD, 0);
  gemm(memH, wvH, b_swv, vswF, nullptr, MROWS, DD, DD, 0);
  banded_attn<<<dim3(TT, BB * HH), 64, 0, stream>>>(qswF, kswF, vswF, projF);
  cvt(projF, oH, MROWS * DD);
  gemm(oH, woH, b_swo, swoF, nullptr, MROWS, DD, DD, 0);
  residual_ln<<<MROWS, 256, 0, stream>>>(x2F, swoF, n2g, n2b, x3F, xH);

  // 4) FFN
  gemm(xH, wl1H, b_l1, nullptr, hH, MROWS, DFFN, DD, 1);
  gemm(hH, wl2H, b_l2, ffF, nullptr, MROWS, DD, DFFN, 0);
  residual_ln<<<MROWS, 256, 0, stream>>>(x3F, ffF, n3g, n3b, (float*)d_out,
                                         nullptr);

  (void)in_sizes; (void)n_in; (void)out_size; (void)ws_size;
}
